// UniGCNConv_29970281792063
// MI455X (gfx1250) — matrix operand order repaired and checked by reading, not compiled, not run on hardware
//
#include <hip/hip_runtime.h>
#include <hip/hip_bf16.h>

// ---------------------------------------------------------------------------
// UniGCNConv for MI455X (gfx1250, wave32, WMMA)
//   Xp = X @ W^T                      (fp32 WMMA 16x16x4)
//   Xe = segmean_edge(Xp[vertex]) * degE      (CSR gather, no float atomics)
//   Xv = segsum_vertex(Xe[edges]) * degV
// ---------------------------------------------------------------------------

#define CCH 256   // IN_C == OUT_C == 256

typedef float v2f  __attribute__((ext_vector_type(2)));
typedef float v8f  __attribute__((ext_vector_type(8)));

#if defined(__has_builtin)
#if __has_builtin(__builtin_amdgcn_wmma_f32_16x16x4_f32)
#define HAVE_WMMA_F32 1
#endif
#endif

// ---------------------------- utility kernels ------------------------------

__global__ void zero_i32_kernel(int* __restrict__ p, int n) {
  int i = blockIdx.x * blockDim.x + threadIdx.x;
  if (i < n) p[i] = 0;
}

// row sums of H (N x M row-major): one block per row, coalesced across columns
__global__ void rowsum_kernel(const float* __restrict__ H, float* __restrict__ sumV, int Me) {
  const int row = blockIdx.x;
  const float* h = H + (size_t)row * Me;
  float acc = 0.f;
  for (int j = threadIdx.x; j < Me; j += blockDim.x) acc += h[j];
  __shared__ float red[8];
  for (int off = 16; off > 0; off >>= 1) acc += __shfl_down(acc, off, 32);
  const int wid = threadIdx.x >> 5;
  if ((threadIdx.x & 31) == 0) red[wid] = acc;
  __syncthreads();
  if (threadIdx.x < 8) {
    acc = red[threadIdx.x];
    for (int off = 4; off > 0; off >>= 1) acc += __shfl_down(acc, off, 32);
    if (threadIdx.x == 0) sumV[row] = acc;
  }
}

// column sums of H: tile of rows per block.y, one column per thread (coalesced)
__global__ void colsum_kernel(const float* __restrict__ H, float* __restrict__ sumE,
                              int Nv, int Me, int rowsPer) {
  const int c = blockIdx.x * blockDim.x + threadIdx.x;
  if (c >= Me) return;
  const int r0 = blockIdx.y * rowsPer;
  const int r1 = min(r0 + rowsPer, Nv);
  float acc = 0.f;
  for (int r = r0; r < r1; ++r) acc += H[(size_t)r * Me + c];
  atomicAdd(&sumE[c], acc);
}

__global__ void hist_kernel(const int* __restrict__ vertex, const int* __restrict__ edges,
                            int* __restrict__ cntV, int* __restrict__ cntE, int nnz) {
  int i = blockIdx.x * blockDim.x + threadIdx.x;
  if (i >= nnz) return;
  atomicAdd(&cntE[edges[i]], 1);
  atomicAdd(&cntV[vertex[i]], 1);
}

// single-block exclusive scan (Hillis-Steele per 1024-chunk + running carry)
__global__ void scan_excl_kernel(const int* __restrict__ in, int* __restrict__ out, int n) {
  __shared__ int buf[1024];
  __shared__ int carry;
  const int t = threadIdx.x;
  if (t == 0) carry = 0;
  __syncthreads();
  for (int base = 0; base < n; base += 1024) {
    const int x = (base + t < n) ? in[base + t] : 0;
    buf[t] = x;
    __syncthreads();
    for (int off = 1; off < 1024; off <<= 1) {
      int v = (t >= off) ? buf[t - off] : 0;
      __syncthreads();
      buf[t] += v;
      __syncthreads();
    }
    const int incl_last = buf[1023];
    if (base + t < n) out[base + t] = carry + buf[t] - x;
    __syncthreads();
    if (t == 0) carry += incl_last;
    __syncthreads();
  }
  if (t == 0) out[n] = carry;
}

__global__ void fill_csr_kernel(const int* __restrict__ vertex, const int* __restrict__ edges,
                                const int* __restrict__ offE, const int* __restrict__ offV,
                                int* __restrict__ curE, int* __restrict__ curV,
                                int* __restrict__ csrE, int* __restrict__ csrV, int nnz) {
  int i = blockIdx.x * blockDim.x + threadIdx.x;
  if (i >= nnz) return;
  const int e = edges[i];
  const int p = atomicAdd(&curE[e], 1);
  csrE[offE[e] + p] = i;
  const int v = vertex[i];
  const int q = atomicAdd(&curV[v], 1);
  csrV[offV[v] + q] = i;
}

__global__ void scales_e_kernel(const float* __restrict__ sumE, const int* __restrict__ cntE,
                                float* __restrict__ scaleE, int Me) {
  int e = blockIdx.x * blockDim.x + threadIdx.x;
  if (e >= Me) return;
  const float s = sumE[e];
  const float d = (s > 0.f) ? (1.0f / sqrtf(s)) : 1.0f;   // inf -> 1 handling
  scaleE[e] = d / ((float)cntE[e] + 1e-8f);               // mean + edge norm folded
}

__global__ void scales_v_kernel(const float* __restrict__ sumV, float* __restrict__ scaleV, int Nv) {
  int v = blockIdx.x * blockDim.x + threadIdx.x;
  if (v >= Nv) return;
  const float s = sumV[v];
  scaleV[v] = (s > 0.f) ? (1.0f / sqrtf(s)) : 1.0f;
}

// ------------------------------- GEMM (WMMA) -------------------------------
// Xp[n][o] = sum_k X[n][k] * W[o][k];  X: [N,256], W: [256,256] both row-major.
// One wave -> 16 rows x 64 cols strip; 4 accumulators share the A fragment.

#if HAVE_WMMA_F32
__global__ void gemm_wmma_kernel(const float* __restrict__ X, const float* __restrict__ W,
                                 float* __restrict__ Xp, int nrows) {
  const int lane = threadIdx.x & 31;
  const int wave = (blockIdx.x * blockDim.x + threadIdx.x) >> 5;
  const int row0 = (wave >> 2) * 16;
  const int colT = (wave & 3) << 6;            // 0,64,128,192
  if (row0 >= nrows) return;                   // wave-uniform: EXEC stays all-ones
  const int m   = lane & 15;
  const int khi = (lane >> 4) << 1;            // lanes 0-15: K+0/1, lanes 16-31: K+2/3
  v8f a0 = {}, a1 = {}, a2 = {}, a3 = {};
  const float* xr = X + (size_t)(row0 + m) * CCH;
  const float* w0 = W + (size_t)(colT + m) * CCH;       // B[k][n] = W[n][k]
  const float* w1 = w0 + 16 * CCH;
  const float* w2 = w0 + 32 * CCH;
  const float* w3 = w0 + 48 * CCH;
  for (int k = 0; k < CCH; k += 4) {
    const v2f a  = *(const v2f*)(xr + k + khi);
    const v2f b0 = *(const v2f*)(w0 + k + khi);
    const v2f b1 = *(const v2f*)(w1 + k + khi);
    const v2f b2 = *(const v2f*)(w2 + k + khi);
    const v2f b3 = *(const v2f*)(w3 + k + khi);
    a0 = __builtin_amdgcn_wmma_f32_16x16x4_f32(false, a, false, b0, (short)0, a0, false, false);
    a1 = __builtin_amdgcn_wmma_f32_16x16x4_f32(false, a, false, b1, (short)0, a1, false, false);
    a2 = __builtin_amdgcn_wmma_f32_16x16x4_f32(false, a, false, b2, (short)0, a2, false, false);
    a3 = __builtin_amdgcn_wmma_f32_16x16x4_f32(false, a, false, b3, (short)0, a3, false, false);
  }
  // D layout: VGPR r -> row r (lanes 0-15) / row r+8 (lanes 16-31), col = lane&15
  const int rbase = row0 + ((lane >> 4) << 3);
  float* op = Xp + (size_t)rbase * CCH + colT + m;
#pragma unroll
  for (int r = 0; r < 8; ++r) {
    float* row = op + (size_t)r * CCH;
    row[0]  = a0[r];
    row[16] = a1[r];
    row[32] = a2[r];
    row[48] = a3[r];
  }
}
#else
// Fallback: probe-confirmed bf16 WMMA with on-the-fly f32->bf16 conversion.
typedef __bf16 bf16_t;
typedef __attribute__((ext_vector_type(16))) __bf16 v16bf;

__global__ void gemm_wmma_kernel(const float* __restrict__ X, const float* __restrict__ W,
                                 float* __restrict__ Xp, int nrows) {
  const int lane = threadIdx.x & 31;
  const int wave = (blockIdx.x * blockDim.x + threadIdx.x) >> 5;
  const int row0 = (wave >> 2) * 16;
  const int colT = (wave & 3) << 6;
  if (row0 >= nrows) return;
  const int m   = lane & 15;
  const int hiA = (lane >> 4) * 8;
  const int hiB = (lane >> 4) * 16;
  v8f a0 = {}, a1 = {}, a2 = {}, a3 = {};
  const float* xr = X + (size_t)(row0 + m) * CCH;
  const float* w0 = W + (size_t)(colT + m) * CCH;
  for (int k = 0; k < CCH; k += 32) {
    v16bf a, b0, b1, b2, b3;
#pragma unroll
    for (int e = 0; e < 16; ++e) {
      const int ka = k + hiA + (e < 8 ? e : e + 8);   // 16-bit A 16x32 layout
      const int kb = k + hiA + (e < 8 ? e : e + 8);   // 16-bit B 32x16 layout
      a[e]  = (bf16_t)xr[ka];
      b0[e] = (bf16_t)w0[kb];
      b1[e] = (bf16_t)w0[16 * CCH + kb];
      b2[e] = (bf16_t)w0[32 * CCH + kb];
      b3[e] = (bf16_t)w0[48 * CCH + kb];
    }
    a0 = __builtin_amdgcn_wmma_f32_16x16x32_bf16(false, a, false, b0, (short)0, a0, false, false);
    a1 = __builtin_amdgcn_wmma_f32_16x16x32_bf16(false, a, false, b1, (short)0, a1, false, false);
    a2 = __builtin_amdgcn_wmma_f32_16x16x32_bf16(false, a, false, b2, (short)0, a2, false, false);
    a3 = __builtin_amdgcn_wmma_f32_16x16x32_bf16(false, a, false, b3, (short)0, a3, false, false);
  }
  const int rbase = row0 + ((lane >> 4) << 3);
  float* op = Xp + (size_t)rbase * CCH + colT + m;
#pragma unroll
  for (int r = 0; r < 8; ++r) {
    float* row = op + (size_t)r * CCH;
    row[0]  = a0[r];
    row[16] = a1[r];
    row[32] = a2[r];
    row[48] = a3[r];
  }
}
#endif

// -------------------------- CSR segment aggregation ------------------------
// one block (256 thr = 1 thr/channel) per segment; rows stream through L2

__global__ void edge_agg_kernel(const float* __restrict__ Xp, const int* __restrict__ csrE,
                                const int* __restrict__ offE, const int* __restrict__ vertex,
                                const float* __restrict__ scaleE, float* __restrict__ Xe) {
  const int e = blockIdx.x, c = threadIdx.x;
  const int p0 = offE[e], p1 = offE[e + 1];
  float acc = 0.f;
  for (int p = p0; p < p1; ++p) acc += Xp[(size_t)vertex[csrE[p]] * CCH + c];
  Xe[(size_t)e * CCH + c] = acc * scaleE[e];
}

__global__ void vert_agg_kernel(const float* __restrict__ Xe, const int* __restrict__ csrV,
                                const int* __restrict__ offV, const int* __restrict__ edges,
                                const float* __restrict__ scaleV, float* __restrict__ out) {
  const int v = blockIdx.x, c = threadIdx.x;
  const int p0 = offV[v], p1 = offV[v + 1];
  float acc = 0.f;
  for (int p = p0; p < p1; ++p) acc += Xe[(size_t)edges[csrV[p]] * CCH + c];
  out[(size_t)v * CCH + c] = acc * scaleV[v];
}

// --------------------------------- launch ----------------------------------

extern "C" void kernel_launch(void* const* d_in, const int* in_sizes, int n_in,
                              void* d_out, int out_size, void* d_ws, size_t ws_size,
                              hipStream_t stream) {
  const float* X      = (const float*)d_in[0];
  const float* W      = (const float*)d_in[1];
  const int*   vertex = (const int*)  d_in[2];
  const int*   edges  = (const int*)  d_in[3];
  const float* H      = (const float*)d_in[4];
  float*       out    = (float*)d_out;

  const int Nv  = in_sizes[0] / CCH;   // 20000
  const int nnz = in_sizes[2];         // 160000
  const int Me  = in_sizes[4] / Nv;    // 4000

  // workspace layout
  float* Xp     = (float*)d_ws;                 // Nv*CCH
  float* Xe     = Xp + (size_t)Nv * CCH;        // Me*CCH
  float* sumV   = Xe + (size_t)Me * CCH;        // Nv
  float* scaleV = sumV + Nv;                    // Nv
  float* scaleE = scaleV + Nv;                  // Me
  int*   offE   = (int*)(scaleE + Me);          // Me+1
  int*   offV   = offE + (Me + 1);              // Nv+1
  int*   csrE   = offV + (Nv + 1);              // nnz
  int*   csrV   = csrE + nnz;                   // nnz
  int*   zbase  = csrV + nnz;                   // start of zero-init region
  float* sumE   = (float*)zbase;                // Me
  int*   cntE   = zbase + Me;                   // Me
  int*   cntV   = cntE + Me;                    // Nv
  int*   curE   = cntV + Nv;                    // Me
  int*   curV   = curE + Me;                    // Nv
  const int nzero = 3 * Me + 2 * Nv;

  // 1) re-init atomic targets (graph-replay safe)
  zero_i32_kernel<<<(nzero + 255) / 256, 256, 0, stream>>>(zbase, nzero);

  // 2) degrees from dense H (HBM-bound, ~2x320MB)
  rowsum_kernel<<<Nv, 256, 0, stream>>>(H, sumV, Me);
  {
    const int rowsPer = 128;
    dim3 g((Me + 255) / 256, (Nv + rowsPer - 1) / rowsPer);
    colsum_kernel<<<g, 256, 0, stream>>>(H, sumE, Nv, Me, rowsPer);
  }

  // 3) COO histograms + CSR build (int atomics only)
  hist_kernel<<<(nnz + 255) / 256, 256, 0, stream>>>(vertex, edges, cntV, cntE, nnz);
  scan_excl_kernel<<<1, 1024, 0, stream>>>(cntE, offE, Me);
  scan_excl_kernel<<<1, 1024, 0, stream>>>(cntV, offV, Nv);
  fill_csr_kernel<<<(nnz + 255) / 256, 256, 0, stream>>>(vertex, edges, offE, offV,
                                                         curE, curV, csrE, csrV, nnz);
  scales_e_kernel<<<(Me + 255) / 256, 256, 0, stream>>>(sumE, cntE, scaleE, Me);
  scales_v_kernel<<<(Nv + 255) / 256, 256, 0, stream>>>(sumV, scaleV, Nv);

  // 4) Xp = X @ W^T via WMMA (one wave per 16x64 strip)
  {
    const int waves  = ((Nv + 15) / 16) * 4;
    const int blocks = (waves + 7) / 8;        // 8 waves / 256-thread block
    gemm_wmma_kernel<<<blocks, 256, 0, stream>>>(X, W, Xp, Nv);
  }

  // 5) vertex->edge mean (+degE), then edge->vertex sum (+degV)
  edge_agg_kernel<<<Me, CCH, 0, stream>>>(Xp, csrE, offE, vertex, scaleE, Xe);
  vert_agg_kernel<<<Nv, CCH, 0, stream>>>(Xe, csrV, offV, edges, scaleV, out);
}
